// Tokenizer_14869176779135
// MI455X (gfx1250) — compile-verified, matrix operand order checked by reading, not run on hardware
//
#include <hip/hip_runtime.h>
#include <hip/hip_bf16.h>

typedef __attribute__((ext_vector_type(16))) __bf16 v16bf;
typedef __attribute__((ext_vector_type(8)))  float  v8f;

// Problem constants (from reference): x is (64,32,256*16) fp32, emb (1024,256) fp32
#define NBL      2048     // b*l = 64*32
#define NROWS    32768    // NBL * 16 query rows
#define KDIM     256
#define VOCAB    1024
#define NCODET   64       // 1024 / 16 code tiles
#define KCHUNKS  8        // 256 / 32
#define TILE_ELTS (KCHUNKS * 32 * 16)   // bf16 elements per packed tile (8KB)

// ---------------------------------------------------------------------------
// Pack codebook (1024x256 f32, row-major) into WMMA 16-bit B-fragment layout.
// Fragment lane layout (ISA 7.12.2, 16-bit A/B 16xK tile, wave32):
//   lanes 0-15 : row m=lane,    K = {k0..k0+7} then {k0+16..k0+23}
//   lanes 16-31: row m=lane-16, K = {k0+8..k0+15} then {k0+24..k0+31}
// Stored as 32B per (tile, kchunk, lane) so the GEMM loop does pure b128 traffic.
// ---------------------------------------------------------------------------
__global__ __launch_bounds__(256) void pack_e_kernel(const float* __restrict__ emb,
                                                     __bf16* __restrict__ eb) {
    const int t     = blockIdx.x;          // code tile 0..63
    const int c     = threadIdx.x >> 5;    // k-chunk 0..7
    const int lane  = threadIdx.x & 31;
    const int m     = lane & 15;           // code within tile
    const int khalf = (lane >> 4) ? 8 : 0;
    const float* row = emb + (size_t)(t * 16 + m) * KDIM + c * 32;
    v16bf f;
#pragma unroll
    for (int j = 0; j < 8; ++j) {
        f[j]     = (__bf16)row[khalf + j];
        f[8 + j] = (__bf16)row[16 + khalf + j];
    }
    *(v16bf*)(eb + ((size_t)(t * KCHUNKS + c) * 32 + lane) * 16) = f;
}

// ||e_j||^2 for all 1024 codes (exact fp32)
__global__ __launch_bounds__(256) void enorm_kernel(const float* __restrict__ emb,
                                                    float* __restrict__ enorm) {
    const int j = blockIdx.x * 256 + threadIdx.x;   // 4 blocks -> 1024 codes
    const float4* row = (const float4*)(emb + (size_t)j * KDIM);
    float s = 0.f;
#pragma unroll 8
    for (int i = 0; i < KDIM / 4; ++i) {
        float4 v = row[i];
        s += v.x * v.x + v.y * v.y + v.z * v.z + v.w * v.w;
    }
    enorm[j] = s;
}

// ---------------------------------------------------------------------------
// Pack z into A-fragment layout. For (b,l)-block bl:
//   z_flat[bl*16 + v, k] = x[bl*4096 + k*16 + v]   (the (a v) interleave)
// One block per bl: stage the 16KB chunk in LDS, emit 8 kchunk fragments.
// ---------------------------------------------------------------------------
__global__ __launch_bounds__(256) void pack_z_kernel(const float* __restrict__ x,
                                                     __bf16* __restrict__ za) {
    __shared__ float tile[KDIM * 16];      // 16 KB
    const int bl = blockIdx.x;
    const float* src = x + (size_t)bl * (KDIM * 16);
#pragma unroll
    for (int i = 0; i < 16; ++i)
        tile[threadIdx.x + i * 256] = src[threadIdx.x + i * 256];
    __syncthreads();

    const int c     = threadIdx.x >> 5;    // k-chunk 0..7
    const int lane  = threadIdx.x & 31;
    const int m     = lane & 15;           // row v within tile
    const int khalf = (lane >> 4) ? 8 : 0;
    v16bf f;
#pragma unroll
    for (int j = 0; j < 8; ++j) {
        const int k0 = c * 32 + khalf + j;
        const int k1 = c * 32 + 16 + khalf + j;
        f[j]     = (__bf16)tile[k0 * 16 + m];
        f[8 + j] = (__bf16)tile[k1 * 16 + m];
    }
    *(v16bf*)(za + ((size_t)(bl * KCHUNKS + c) * 32 + lane) * 16) = f;
}

// ---------------------------------------------------------------------------
// Main kernel: per-wave 16-row tile vs all 1024 codes.
// dist-equivalent = ||e||^2 - 2 * (z . e)   (row norm constant -> dropped)
// Codebook tiles stream global->LDS with CDNA5 async copies (ASYNCcnt),
// triple-buffered: one barrier per tile, prefetch 2 tiles ahead, exactly one
// tile copy in flight (s_wait_asynccnt 2).
// C layout: VGPR r, lanes 0-15 -> (M=r,   N=lane)
//                   lanes 16-31 -> (M=8+r, N=lane-16)
// ---------------------------------------------------------------------------
__global__ __launch_bounds__(256) void vq_argmin_kernel(const __bf16* __restrict__ za,
                                                        const __bf16* __restrict__ eb,
                                                        const float*  __restrict__ enorm,
                                                        int*   __restrict__ tokens,
                                                        float* __restrict__ tok_out) {
    __shared__ __bf16 bbuf[3][TILE_ELTS];  // 3 x 8KB rotating buffers

    const int tid  = threadIdx.x;
    const int lane = tid & 31;
    const int wid  = tid >> 5;
    const int rt   = blockIdx.x * 8 + wid;          // row tile 0..2047

    // Each thread owns a 32-byte slice of each 8KB tile copy.
    unsigned ldsoff[3];
#pragma unroll
    for (int i = 0; i < 3; ++i)
        ldsoff[i] = (unsigned)(size_t)(&bbuf[i][0]) + (unsigned)tid * 32u;
    const char* gslice = (const char*)eb + (size_t)tid * 32;

    // Async copy of one tile slice (2 x b128 per lane; same imm offset applies
    // to both LDS and global sides per ISA 10.x async addressing).
    auto issue_tile = [&](int tile, unsigned ldst) {
        unsigned long long g =
            (unsigned long long)(size_t)(gslice + (size_t)tile * (TILE_ELTS * 2));
        asm volatile("global_load_async_to_lds_b128 %0, %1, off"
                     :: "v"(ldst), "v"(g) : "memory");
        asm volatile("global_load_async_to_lds_b128 %0, %1, off offset:16"
                     :: "v"(ldst), "v"(g) : "memory");
    };

    issue_tile(0, ldsoff[0]);
    issue_tile(1, ldsoff[1]);

    // A fragments resident for the whole code loop (64 VGPRs)
    v16bf a[KCHUNKS];
    const v16bf* aptr = (const v16bf*)(za + (size_t)rt * TILE_ELTS);
#pragma unroll
    for (int c = 0; c < KCHUNKS; ++c) a[c] = aptr[c * 32 + lane];

    float best[8];
    int   bidx[8];
#pragma unroll
    for (int r = 0; r < 8; ++r) { best[r] = 3.4e38f; bidx[r] = 0; }

    const int ncol = lane & 15;
    int bi = 0;                               // t % 3
    for (int t = 0; t < NCODET; ++t) {
        // Tile t complete (leave the single in-flight prefetch outstanding).
        if (t < NCODET - 1) asm volatile("s_wait_asynccnt 2" ::: "memory");
        else                asm volatile("s_wait_asynccnt 0" ::: "memory");
        __syncthreads();   // tile t visible; everyone done reading buf[(t+2)%3]

        int pi = bi + 2; if (pi >= 3) pi -= 3;
        if (t + 2 < NCODET) issue_tile(t + 2, ldsoff[pi]);

        // Load all 8 B fragments as one DS clause and pin them live so the
        // scheduler cannot sink loads between WMMAs (kills per-WMMA dscnt stalls).
        const v16bf* bptr = (const v16bf*)(&bbuf[0][0] + (size_t)bi * TILE_ELTS);
        v16bf b[KCHUNKS];
#pragma unroll
        for (int c = 0; c < KCHUNKS; ++c) b[c] = bptr[c * 32 + lane];
        asm volatile("" : "+v"(b[0]), "+v"(b[1]), "+v"(b[2]), "+v"(b[3]),
                          "+v"(b[4]), "+v"(b[5]), "+v"(b[6]), "+v"(b[7]));

        v8f acc = {};
#pragma unroll
        for (int c = 0; c < KCHUNKS; ++c) {
            acc = __builtin_amdgcn_wmma_f32_16x16x32_bf16(
                false, a[c], false, b[c], (short)0, acc, false, false);
        }

        const int   code = t * 16 + ncol;
        const float en   = enorm[code];
#pragma unroll
        for (int r = 0; r < 8; ++r) {
            const float d = en - 2.0f * acc[r];
            if (d < best[r]) { best[r] = d; bidx[r] = code; }
        }
        if (++bi == 3) bi = 0;
    }

    // argmin across the 16 lanes sharing each row group (wave32 xor shuffles)
#pragma unroll
    for (int s = 1; s < 16; s <<= 1) {
#pragma unroll
        for (int r = 0; r < 8; ++r) {
            const float ob = __shfl_xor(best[r], s, 32);
            const int   oi = __shfl_xor(bidx[r], s, 32);
            if (ob < best[r] || (ob == best[r] && oi < bidx[r])) {
                best[r] = ob; bidx[r] = oi;
            }
        }
    }
    if (ncol == 0) {
        const int rowbase = rt * 16 + ((lane >> 4) ? 8 : 0);
#pragma unroll
        for (int r = 0; r < 8; ++r) {
            tokens[rowbase + r]  = bidx[r];
            tok_out[rowbase + r] = (float)bidx[r];   // tokens as fp32 in d_out
        }
    }
}

// ---------------------------------------------------------------------------
// Gather + straight-through output. One block per (b,l); thread t owns dim a=t
// and writes the 16 v-values contiguously (64B/thread, fully coalesced).
// z_q[bl, a, v]         = emb[tok[v], a]
// decoder_input[bl,a,v] = x + ((z_q) - x)   (faithful fp32 rounding)
// ---------------------------------------------------------------------------
__global__ __launch_bounds__(256) void gather_kernel(const float* __restrict__ emb,
                                                     const float* __restrict__ x,
                                                     const int*   __restrict__ tokens,
                                                     float* __restrict__ zq,
                                                     float* __restrict__ dec) {
    __shared__ int tok[16];
    const int bl = blockIdx.x;
    if (threadIdx.x < 16) tok[threadIdx.x] = tokens[bl * 16 + threadIdx.x];
    __syncthreads();

    const int a = threadIdx.x;
    float4 q[4], d[4];
    float* qv = (float*)q;
    float* dv = (float*)d;
    const float* xp = x + (size_t)bl * (KDIM * 16) + (size_t)a * 16;
#pragma unroll
    for (int v = 0; v < 16; ++v) {
        const float e  = emb[(size_t)tok[v] * KDIM + a];  // coalesced across threads
        const float xv = xp[v];
        qv[v] = e;
        dv[v] = xv + (e - xv);
    }
    float4* dq = (float4*)(zq  + ((size_t)bl * KDIM + a) * 16);
    float4* dd = (float4*)(dec + ((size_t)bl * KDIM + a) * 16);
#pragma unroll
    for (int i = 0; i < 4; ++i) { dq[i] = q[i]; dd[i] = d[i]; }
}

extern "C" void kernel_launch(void* const* d_in, const int* in_sizes, int n_in,
                              void* d_out, int out_size, void* d_ws, size_t ws_size,
                              hipStream_t stream) {
    const float* x   = (const float*)d_in[0];   // 64*32*4096 fp32
    const float* emb = (const float*)d_in[1];   // 1024*256 fp32
    float* out = (float*)d_out;

    // d_out: [z_q | decoder_input | tokens-as-f32]
    float* zq      = out;
    float* dec     = out + (size_t)NROWS * KDIM;
    float* tok_out = out + (size_t)2 * NROWS * KDIM;

    // workspace partition (≈17 MB)
    char* ws = (char*)d_ws;
    __bf16* za    = (__bf16*)ws;                                   // 16 MB
    __bf16* eb    = (__bf16*)(ws + (size_t)NROWS * KDIM * 2);      // 512 KB
    float*  enorm = (float*)(ws + (size_t)NROWS * KDIM * 2
                                + (size_t)VOCAB * KDIM * 2);       // 4 KB
    int*    toks  = (int*)((char*)enorm + VOCAB * sizeof(float));  // 128 KB

    pack_e_kernel  <<<NCODET,    256, 0, stream>>>(emb, eb);
    enorm_kernel   <<<VOCAB/256, 256, 0, stream>>>(emb, enorm);
    pack_z_kernel  <<<NBL,       256, 0, stream>>>(x, za);
    vq_argmin_kernel<<<NBL/8,    256, 0, stream>>>(za, eb, enorm, toks, tok_out);
    gather_kernel  <<<NBL,       256, 0, stream>>>(emb, x, toks, zq, dec);
}
